// SelfAttention_29308856828362
// MI455X (gfx1250) — matrix operand order repaired and checked by reading, not compiled, not run on hardware
//
#include <hip/hip_runtime.h>

// ---------------------------------------------------------------------------
// Self-attention forward for MI455X (gfx1250, wave32, WMMA bf16 path).
//   qkv = x @ w_in + b_in ; attn = softmax(q k^T / sqrt(dh)) v ; y = attn @ w_out + b_out
// All GEMMs run on v_wmma_f32_16x16x32_bf16 with fp32 accumulation.
// Round-3: ping-pong (2x unrolled) k-loops -- removes the v_dual_mov_b32
// rotation copies the round-2 double-buffer produced; attention double-buffers
// whole key blocks so score WMMAs never wait on K loads.
// ---------------------------------------------------------------------------

typedef __attribute__((ext_vector_type(16))) __bf16 v16bf;
typedef __attribute__((ext_vector_type(8)))  float  v8f;

#define N_HEADS 16
#define D_EMBED 1024
#define HEAD_SZ 64
#define BATCH   4
#define SEQ     2048
#define ROWS    (BATCH * SEQ)     /* 8192 */
#define QKV_N   (3 * D_EMBED)     /* 3072 */

__device__ __forceinline__ __bf16 to_bf16(float f) { return (__bf16)f; }

__device__ __forceinline__ v8f vzero8() {
  v8f v;
#pragma unroll
  for (int i = 0; i < 8; ++i) v[i] = 0.0f;
  return v;
}

__device__ __forceinline__ v8f wmma_bf16(v16bf a, v16bf b, v8f c) {
  return __builtin_amdgcn_wmma_f32_16x16x32_bf16(
      /*neg_a=*/false, a, /*neg_b=*/false, b,
      /*c_mod=*/(short)0, c, /*reuse_a=*/false, /*reuse_b=*/false);
}

// A fragment: 16x32 (MxK) bf16, row-major source with leading dim lda.
// Lanes 0-15: row M=lane, K = k0+0..7 / k0+16..23; lanes 16-31: +8 / +24.
__device__ __forceinline__ v16bf load_a_frag(const __bf16* A, size_t lda,
                                             int row0, int k0) {
  const int lane = threadIdx.x & 31;
  const int m    = lane & 15;
  const int kb   = k0 + ((lane >> 4) << 3);
  const __bf16* p = A + (size_t)(row0 + m) * lda + kb;
  union { v16bf v; uint4 q[2]; } u;
  u.q[0] = *(const uint4*)(p);        // K = kb .. kb+7
  u.q[1] = *(const uint4*)(p + 16);   // K = kb+16 .. kb+23
  return u.v;
}

// B fragment: 32x16 (KxN) bf16 from an N-major buffer Bt[n, k] (ld = ldb).
// Lanes 0-15: column N=n0+lane, K = k0+0..7 / k0+16..23; lanes 16-31: +8 / +24.
__device__ __forceinline__ v16bf load_b_frag(const __bf16* Bt, size_t ldb,
                                             int n0, int k0) {
  const int lane = threadIdx.x & 31;
  const int n    = n0 + (lane & 15);
  const int kb   = k0 + ((lane >> 4) << 3);
  const __bf16* p = Bt + (size_t)n * ldb + kb;
  union { v16bf v; uint4 q[2]; } u;
  u.q[0] = *(const uint4*)(p);
  u.q[1] = *(const uint4*)(p + 16);
  return u.v;
}

// ---------------------------------------------------------------------------
// Precision conversion kernels
// ---------------------------------------------------------------------------
__global__ void cvt_bf16_kernel(const float* __restrict__ in,
                                __bf16* __restrict__ out, int n) {
  int i = blockIdx.x * blockDim.x + threadIdx.x;
  if (i < n) out[i] = to_bf16(in[i]);
}

// in[rows, cols] (row-major f32) -> out[cols, rows] (row-major bf16)
__global__ void transpose_bf16_kernel(const float* __restrict__ in,
                                      __bf16* __restrict__ out,
                                      int rows, int cols) {
  int i = blockIdx.x * blockDim.x + threadIdx.x;
  if (i < rows * cols) {
    int r = i / cols, c = i % cols;
    out[(size_t)c * rows + r] = to_bf16(in[i]);
  }
}

// ---------------------------------------------------------------------------
// QKV projection: (8192 x 1024) @ (1024 x 3072) + b_in.
// One wave -> 32x64 output tile, k-loop unrolled 2x with ping-pong stages.
// Epilogue scatters into q/k/v buffers:
//   q : (B,H,S,Dh) bf16, pre-scaled by 1/sqrt(Dh)
//   k : (B,H,S,Dh) bf16
//   vT: (B,H,Dh,S) bf16   (transposed so PV B-fragments are contiguous)
// ---------------------------------------------------------------------------
__global__ __launch_bounds__(256) void qkv_gemm_kernel(
    const __bf16* __restrict__ xb, const __bf16* __restrict__ wT,
    const float* __restrict__ b_in, __bf16* __restrict__ qout,
    __bf16* __restrict__ kout, __bf16* __restrict__ vtout) {
  const int lane = threadIdx.x & 31;
  const int wid  = (blockIdx.x * blockDim.x + threadIdx.x) >> 5;
  const int tilesN = QKV_N / 64;           // 48
  const int tn = wid % tilesN;
  const int tm = wid / tilesN;             // < 256
  if (tm >= ROWS / 32) return;
  const int row0 = tm * 32, ncol0 = tn * 64;

  v8f acc0[4], acc1[4];
#pragma unroll
  for (int j = 0; j < 4; ++j) { acc0[j] = vzero8(); acc1[j] = vzero8(); }

  // stage A: k = 0
  v16bf a0A = load_a_frag(xb, D_EMBED, row0,      0);
  v16bf a1A = load_a_frag(xb, D_EMBED, row0 + 16, 0);
  v16bf bfA[4];
#pragma unroll
  for (int j = 0; j < 4; ++j) bfA[j] = load_b_frag(wT, D_EMBED, ncol0 + j * 16, 0);

  for (int k0 = 0; k0 < D_EMBED; k0 += 64) {
    const int k1 = k0 + 32;
    // stage B loads (overlap stage-A WMMAs)
    v16bf a0B = load_a_frag(xb, D_EMBED, row0,      k1);
    v16bf a1B = load_a_frag(xb, D_EMBED, row0 + 16, k1);
    v16bf bfB[4];
#pragma unroll
    for (int j = 0; j < 4; ++j) bfB[j] = load_b_frag(wT, D_EMBED, ncol0 + j * 16, k1);
    __builtin_prefetch(xb + (size_t)row0 * D_EMBED + k0 + 128, 0, 3);

#pragma unroll
    for (int j = 0; j < 4; ++j) acc0[j] = wmma_bf16(a0A, bfA[j], acc0[j]);
#pragma unroll
    for (int j = 0; j < 4; ++j) acc1[j] = wmma_bf16(a1A, bfA[j], acc1[j]);

    // stage A loads for next iteration (wrap keeps it branch-free)
    const int k2 = (k0 + 64 < D_EMBED) ? (k0 + 64) : 0;
    a0A = load_a_frag(xb, D_EMBED, row0,      k2);
    a1A = load_a_frag(xb, D_EMBED, row0 + 16, k2);
#pragma unroll
    for (int j = 0; j < 4; ++j) bfA[j] = load_b_frag(wT, D_EMBED, ncol0 + j * 16, k2);

#pragma unroll
    for (int j = 0; j < 4; ++j) acc0[j] = wmma_bf16(a0B, bfB[j], acc0[j]);
#pragma unroll
    for (int j = 0; j < 4; ++j) acc1[j] = wmma_bf16(a1B, bfB[j], acc1[j]);
  }

  const float qscale = 0.125f;  // 1/sqrt(HEAD_SZ)
#pragma unroll
  for (int ms = 0; ms < 2; ++ms) {
#pragma unroll
    for (int r = 0; r < 8; ++r) {
      int m  = row0 + ms * 16 + r + ((lane >> 4) << 3);
      int bb = m >> 11;            // m / SEQ
      int s  = m & (SEQ - 1);
#pragma unroll
      for (int j = 0; j < 4; ++j) {
        int n = ncol0 + j * 16 + (lane & 15);
        float v = (ms ? acc1[j][r] : acc0[j][r]) + b_in[n];
        int sec = n >> 10;          // 0:q 1:k 2:v
        int c   = n & (D_EMBED - 1);
        int h   = c >> 6, dh = c & 63;
        size_t bh = (size_t)(bb * N_HEADS + h);
        if (sec == 0)
          qout[(bh * SEQ + s) * HEAD_SZ + dh] = to_bf16(v * qscale);
        else if (sec == 1)
          kout[(bh * SEQ + s) * HEAD_SZ + dh] = to_bf16(v);
        else
          vtout[(bh * HEAD_SZ + dh) * SEQ + s] = to_bf16(v);
      }
    }
  }
}

// ---------------------------------------------------------------------------
// Flash-style attention: one wave owns a 16-row q tile of one (b,h) problem,
// streams keys in blocks of 32 with online softmax. Key blocks are ping-pong
// double-buffered: block i+1's K/V fragments load while block i is processed,
// so neither score nor PV WMMAs wait on memory. Probs are re-laid-out from
// C-fragment form to A-fragment form through a private LDS tile.
// ---------------------------------------------------------------------------
__global__ __launch_bounds__(256) void attn_kernel(
    const __bf16* __restrict__ qb, const __bf16* __restrict__ kb,
    const __bf16* __restrict__ vtb, __bf16* __restrict__ attn) {
  __shared__ __bf16 smem[8][16 * 40];   // 8 waves/block, 16x32 tile, pad to 40
  const int lane = threadIdx.x & 31;
  const int wv   = threadIdx.x >> 5;
  const int wid  = (blockIdx.x * blockDim.x + threadIdx.x) >> 5;
  const int bh   = wid >> 7;            // 64 (b,h) problems
  const int qt   = wid & 127;           // 128 q tiles of 16 rows
  if (bh >= BATCH * N_HEADS) return;

  const __bf16* Q  = qb  + (size_t)bh * SEQ * HEAD_SZ;
  const __bf16* K  = kb  + (size_t)bh * SEQ * HEAD_SZ;
  const __bf16* Vt = vtb + (size_t)bh * HEAD_SZ * SEQ;
  const int q0 = qt * 16;

  // Q fragments (16x32 each) held in registers for the whole sweep.
  v16bf aq0 = load_a_frag(Q, HEAD_SZ, q0, 0);
  v16bf aq1 = load_a_frag(Q, HEAD_SZ, q0, 32);

  v8f m_run, l_run, o_acc[4];
#pragma unroll
  for (int r = 0; r < 8; ++r) { m_run[r] = -1e30f; l_run[r] = 0.0f; }
#pragma unroll
  for (int j = 0; j < 4; ++j) o_acc[j] = vzero8();

  __bf16* sm = smem[wv];

  // load one 32-key block's K (4) and Vt (4) fragments
  auto load_blk = [&](v16bf* kf, v16bf* vf, int kb0) {
    kf[0] = load_b_frag(K, HEAD_SZ, kb0,      0);
    kf[1] = load_b_frag(K, HEAD_SZ, kb0,      32);
    kf[2] = load_b_frag(K, HEAD_SZ, kb0 + 16, 0);
    kf[3] = load_b_frag(K, HEAD_SZ, kb0 + 16, 32);
#pragma unroll
    for (int j = 0; j < 4; ++j) vf[j] = load_b_frag(Vt, SEQ, j * 16, kb0);
  };

  // process one 32-key block: scores -> online softmax -> LDS relayout -> PV
  auto process = [&](const v16bf* kf, const v16bf* vf) {
    v8f s0 = vzero8(), s1 = vzero8();
    s0 = wmma_bf16(aq0, kf[0], s0);
    s0 = wmma_bf16(aq1, kf[1], s0);
    s1 = wmma_bf16(aq0, kf[2], s1);
    s1 = wmma_bf16(aq1, kf[3], s1);

#pragma unroll
    for (int r = 0; r < 8; ++r) {
      float bmx = fmaxf(s0[r], s1[r]);
      bmx = fmaxf(bmx, __shfl_xor(bmx, 1, 32));
      bmx = fmaxf(bmx, __shfl_xor(bmx, 2, 32));
      bmx = fmaxf(bmx, __shfl_xor(bmx, 4, 32));
      bmx = fmaxf(bmx, __shfl_xor(bmx, 8, 32));
      float mn    = fmaxf(m_run[r], bmx);
      float alpha = __expf(m_run[r] - mn);
      float p0    = __expf(s0[r] - mn);
      float p1    = __expf(s1[r] - mn);
      float rs = p0 + p1;
      rs += __shfl_xor(rs, 1, 32);
      rs += __shfl_xor(rs, 2, 32);
      rs += __shfl_xor(rs, 4, 32);
      rs += __shfl_xor(rs, 8, 32);
      l_run[r] = l_run[r] * alpha + rs;
      m_run[r] = mn;
      s0[r] = p0; s1[r] = p1;
      o_acc[0][r] *= alpha; o_acc[1][r] *= alpha;
      o_acc[2][r] *= alpha; o_acc[3][r] *= alpha;
    }

    // probs: C-fragment layout -> row-major bf16 tile in LDS
    {
      int rowoff = (lane >> 4) << 3;
      int cn     = lane & 15;
#pragma unroll
      for (int r = 0; r < 8; ++r) {
        sm[(r + rowoff) * 40 + cn]      = to_bf16(s0[r]);
        sm[(r + rowoff) * 40 + 16 + cn] = to_bf16(s1[r]);
      }
    }
    asm volatile("s_wait_dscnt 0x0" ::: "memory");  // DS RAW inside the wave

    // reload probs as an A fragment (16x32 over key positions)
    v16bf ap;
    {
      int m_ = lane & 15;
      int kk = (lane >> 4) << 3;
      const __bf16* p = sm + m_ * 40 + kk;
      union { v16bf v; uint4 q[2]; } u;
      u.q[0] = *(const uint4*)(p);
      u.q[1] = *(const uint4*)(p + 16);
      ap = u.v;
    }
    asm volatile("s_wait_dscnt 0x0" ::: "memory");  // DS WAR vs next block

#pragma unroll
    for (int j = 0; j < 4; ++j)
      o_acc[j] = wmma_bf16(ap, vf[j], o_acc[j]);
  };

  v16bf kfA[4], vfA[4], kfB[4], vfB[4];
  load_blk(kfA, vfA, 0);

  for (int kb0 = 0; kb0 < SEQ; kb0 += 64) {
    load_blk(kfB, vfB, kb0 + 32);
    __builtin_prefetch(K + (size_t)(kb0 + 64) * HEAD_SZ, 0, 3);
    process(kfA, vfA);

    const int kn = (kb0 + 64 < SEQ) ? (kb0 + 64) : 0;  // wrap: branch-free
    load_blk(kfA, vfA, kn);
    process(kfB, vfB);
  }

  // ---- normalize and store attn output as (B,S,C) bf16 ----
  const int b_ = bh >> 4, h = bh & 15;
#pragma unroll
  for (int r = 0; r < 8; ++r) {
    float inv = 1.0f / l_run[r];
    int s = q0 + r + ((lane >> 4) << 3);
    size_t rowbase = ((size_t)(b_ * SEQ + s)) * D_EMBED + h * HEAD_SZ;
#pragma unroll
    for (int j = 0; j < 4; ++j)
      attn[rowbase + j * 16 + (lane & 15)] = to_bf16(o_acc[j][r] * inv);
  }
}

// ---------------------------------------------------------------------------
// Output projection: (8192 x 1024) @ (1024 x 1024) + b_out -> f32 d_out.
// Same 32x64 wave tile + ping-pong k-loop as the QKV GEMM.
// ---------------------------------------------------------------------------
__global__ __launch_bounds__(256) void out_gemm_kernel(
    const __bf16* __restrict__ attn, const __bf16* __restrict__ wT,
    const float* __restrict__ b_out, float* __restrict__ out) {
  const int lane = threadIdx.x & 31;
  const int wid  = (blockIdx.x * blockDim.x + threadIdx.x) >> 5;
  const int tilesN = D_EMBED / 64;          // 16
  const int tn = wid % tilesN;
  const int tm = wid / tilesN;              // < 256
  if (tm >= ROWS / 32) return;
  const int row0 = tm * 32, ncol0 = tn * 64;

  v8f acc0[4], acc1[4];
#pragma unroll
  for (int j = 0; j < 4; ++j) { acc0[j] = vzero8(); acc1[j] = vzero8(); }

  v16bf a0A = load_a_frag(attn, D_EMBED, row0,      0);
  v16bf a1A = load_a_frag(attn, D_EMBED, row0 + 16, 0);
  v16bf bfA[4];
#pragma unroll
  for (int j = 0; j < 4; ++j) bfA[j] = load_b_frag(wT, D_EMBED, ncol0 + j * 16, 0);

  for (int k0 = 0; k0 < D_EMBED; k0 += 64) {
    const int k1 = k0 + 32;
    v16bf a0B = load_a_frag(attn, D_EMBED, row0,      k1);
    v16bf a1B = load_a_frag(attn, D_EMBED, row0 + 16, k1);
    v16bf bfB[4];
#pragma unroll
    for (int j = 0; j < 4; ++j) bfB[j] = load_b_frag(wT, D_EMBED, ncol0 + j * 16, k1);
    __builtin_prefetch(attn + (size_t)row0 * D_EMBED + k0 + 128, 0, 3);

#pragma unroll
    for (int j = 0; j < 4; ++j) acc0[j] = wmma_bf16(a0A, bfA[j], acc0[j]);
#pragma unroll
    for (int j = 0; j < 4; ++j) acc1[j] = wmma_bf16(a1A, bfA[j], acc1[j]);

    const int k2 = (k0 + 64 < D_EMBED) ? (k0 + 64) : 0;
    a0A = load_a_frag(attn, D_EMBED, row0,      k2);
    a1A = load_a_frag(attn, D_EMBED, row0 + 16, k2);
#pragma unroll
    for (int j = 0; j < 4; ++j) bfA[j] = load_b_frag(wT, D_EMBED, ncol0 + j * 16, k2);

#pragma unroll
    for (int j = 0; j < 4; ++j) acc0[j] = wmma_bf16(a0B, bfB[j], acc0[j]);
#pragma unroll
    for (int j = 0; j < 4; ++j) acc1[j] = wmma_bf16(a1B, bfB[j], acc1[j]);
  }

#pragma unroll
  for (int ms = 0; ms < 2; ++ms) {
#pragma unroll
    for (int r = 0; r < 8; ++r) {
      int m = row0 + ms * 16 + r + ((lane >> 4) << 3);
#pragma unroll
      for (int j = 0; j < 4; ++j) {
        int n = ncol0 + j * 16 + (lane & 15);
        out[(size_t)m * D_EMBED + n] = (ms ? acc1[j][r] : acc0[j][r]) + b_out[n];
      }
    }
  }
}

// ---------------------------------------------------------------------------
// Host launcher
// ---------------------------------------------------------------------------
extern "C" void kernel_launch(void* const* d_in, const int* in_sizes, int n_in,
                              void* d_out, int out_size, void* d_ws, size_t ws_size,
                              hipStream_t stream) {
  const float* x     = (const float*)d_in[0];
  const float* w_in  = (const float*)d_in[1];
  const float* b_in  = (const float*)d_in[2];
  const float* w_out = (const float*)d_in[3];
  const float* b_out = (const float*)d_in[4];
  float* out = (float*)d_out;

  char* ws = (char*)d_ws;
  size_t off = 0;
  auto carve = [&](size_t bytes) -> char* {
    char* p = ws + off;
    off = (off + bytes + 255) & ~(size_t)255;
    return p;
  };
  __bf16* xb    = (__bf16*)carve((size_t)ROWS * D_EMBED * 2);     // 16 MB
  __bf16* w_inT = (__bf16*)carve((size_t)QKV_N * D_EMBED * 2);    //  6 MB
  __bf16* w_oT  = (__bf16*)carve((size_t)D_EMBED * D_EMBED * 2);  //  2 MB
  __bf16* qbuf  = (__bf16*)carve((size_t)ROWS * D_EMBED * 2);     // 16 MB
  __bf16* kbuf  = (__bf16*)carve((size_t)ROWS * D_EMBED * 2);     // 16 MB
  __bf16* vtbuf = (__bf16*)carve((size_t)ROWS * D_EMBED * 2);     // 16 MB
  __bf16* attnb = (__bf16*)carve((size_t)ROWS * D_EMBED * 2);     // 16 MB
  (void)ws_size; (void)in_sizes; (void)n_in; (void)out_size;

  // 1) precision conversions (bf16, weights transposed to N-major)
  {
    int n = ROWS * D_EMBED;
    cvt_bf16_kernel<<<(n + 255) / 256, 256, 0, stream>>>(x, xb, n);
  }
  {
    int n = D_EMBED * QKV_N;
    transpose_bf16_kernel<<<(n + 255) / 256, 256, 0, stream>>>(w_in, w_inT,
                                                               D_EMBED, QKV_N);
  }
  {
    int n = D_EMBED * D_EMBED;
    transpose_bf16_kernel<<<(n + 255) / 256, 256, 0, stream>>>(w_out, w_oT,
                                                               D_EMBED, D_EMBED);
  }

  // 2) QKV projection: 256 M-tiles x 48 N-tiles = 12288 waves / 8 per block
  qkv_gemm_kernel<<<1536, 256, 0, stream>>>(xb, w_inT, b_in, qbuf, kbuf, vtbuf);

  // 3) attention: 64 (b,h) x 128 q-tiles = 8192 waves / 8 per block
  attn_kernel<<<1024, 256, 0, stream>>>(qbuf, kbuf, vtbuf, attnb);

  // 4) output projection: 256 x 16 tiles = 4096 waves / 8 per block
  out_gemm_kernel<<<512, 256, 0, stream>>>(attnb, w_oT, b_out, out);
}
